// MultiDirectionAdaptiveSSM2DRefiner_2173253452150
// MI455X (gfx1250) — matrix operand order repaired and checked by reading, not compiled, not run on hardware
//
#include <hip/hip_runtime.h>
#include <hip/hip_bf16.h>
#include <math.h>

typedef __attribute__((ext_vector_type(16))) _Float16 v16h;
typedef __attribute__((ext_vector_type(8)))  float    v8f;
typedef _Float16 f16;

// Problem geometry (fixed by the reference).
constexpr int BB = 8, CCH = 256, HHH = 64, WWW = 64;
constexpr int LL = 64;
constexpr int NL = 32768;       // B*H*W sequence-positions per direction
constexpr int SITES = 32768;    // B*H*W spatial sites

// ---------------- workspace layout (bytes) ----------------
constexpr size_t OFF_LAMBR = 0;                       // 4*256 f32 lam_bar re
constexpr size_t OFF_LAMBI = OFF_LAMBR + 4096;        // lam_bar im
constexpr size_t OFF_COEFR = OFF_LAMBI + 4096;        // (lam_bar-1)/lam re
constexpr size_t OFF_COEFI = OFF_COEFR + 4096;        // im
constexpr size_t OFF_CMASK = OFF_COEFI + 4096;        // bandlimit mask
constexpr size_t OFF_MEAN  = OFF_CMASK + 4096;        // 32768 f32 per-site mean
constexpr size_t OFF_RSTD  = OFF_MEAN  + 131072;      // 32768 f32 per-site rstd
constexpr size_t OFF_WBU   = OFF_RSTD  + 131072;      // 4 x [512,256] f16 packed B_bar (re||im rows)
constexpr size_t OFF_WC    = OFF_WBU   + 1048576;     // 4 x [256,512] f16 packed 2C (re || -im cols)
constexpr size_t OFF_WENC  = OFF_WC    + 1048576;     // 4 x [256,256] f16
constexpr size_t OFF_WDEC  = OFF_WENC  + 524288;      // 4 x [256,256] f16
constexpr size_t OFF_WFUSE = OFF_WDEC  + 524288;      // [256,1024] f16
constexpr size_t OFF_FX16  = OFF_WFUSE + 524288;      // [32768,256] f16 (fx, later fx2)
constexpr size_t OFF_FXF   = OFF_FX16  + 16777216;    // [32768,256] f32 (fx, later fx2)
constexpr size_t OFF_BU16  = OFF_FXF   + 33554432;    // [32768,512] f16 Bu
constexpr size_t OFF_XS16  = OFF_BU16  + 33554432;    // [32768,512] f16 xs (later hff16)
constexpr size_t OFF_YS    = OFF_XS16  + 33554432;    // [32768,256] f32 y
constexpr size_t OFF_YALL  = OFF_YS    + 33554432;    // [32768,1024] f16 fused dirs

// ---------------- discretization: lam_bar, (lam_bar-1)/lam, cmask ----------------
__global__ void k_disc(const float* __restrict__ lam_re, const float* __restrict__ lam_im,
                       const float* __restrict__ log_step,
                       float* lbr, float* lbi, float* cr, float* ci, float* cm) {
  int idx = blockIdx.x * 256 + threadIdx.x;      // 4*256
  float lr = lam_re[idx], li = lam_im[idx];
  float st = __expf(log_step[idx]);
  float er = __expf(lr * st);
  float s, c; __sincosf(li * st, &s, &c);
  float br = er * c, bi = er * s;
  lbr[idx] = br; lbi[idx] = bi;
  float den = lr * lr + li * li;
  cr[idx] = ((br - 1.0f) * lr + bi * li) / den;
  ci[idx] = (bi * lr - (br - 1.0f) * li) / den;
  float freq = st * fabsf(li) * 0.15915494309189535f; // /(2*pi)
  cm[idx] = (freq < 0.25f) ? 1.0f : 0.0f;
}

// Pack B_bar = coef * (B_re + i B_im) as f16 [d][row=p(re)|256+p(im)][c]
__global__ void k_wbu(const float* __restrict__ B_re, const float* __restrict__ B_im,
                      const float* __restrict__ cr, const float* __restrict__ ci,
                      f16* __restrict__ wbu) {
  int idx = blockIdx.x * 256 + threadIdx.x;      // 4*256*256
  int d = idx >> 16, p = (idx >> 8) & 255, c = idx & 255;
  int dp = d * 256 + p;
  float br = B_re[idx], bi = B_im[idx];
  float crr = cr[dp], cii = ci[dp];
  size_t base = (size_t)d * 131072;
  wbu[base + (size_t)p * 256 + c]         = (f16)(crr * br - cii * bi);
  wbu[base + (size_t)(256 + p) * 256 + c] = (f16)(crr * bi + cii * br);
}

// Pack Wc[d][c][k]: k<256 -> 2*C_re*mask ; k>=256 -> -2*C_im*mask
__global__ void k_wc(const float* __restrict__ C_re, const float* __restrict__ C_im,
                     const float* __restrict__ cm, f16* __restrict__ wc) {
  int idx = blockIdx.x * 256 + threadIdx.x;      // 4*256*256
  int d = idx >> 16, c = (idx >> 8) & 255, p = idx & 255;
  float m = cm[d * 256 + p];
  size_t src = (size_t)(d * 256 + c) * 256 + p;
  size_t base = (size_t)d * 131072 + (size_t)c * 512;
  wc[base + p]       = (f16)( 2.0f * C_re[src] * m);
  wc[base + 256 + p] = (f16)(-2.0f * C_im[src] * m);
}

__global__ void k_cvt16(const float* __restrict__ src, f16* __restrict__ dst) {
  int idx = blockIdx.x * 256 + threadIdx.x;
  dst[idx] = (f16)src[idx];
}

// Per-site LN1 statistics over channels (direction-invariant).
__global__ void k_stats(const float* __restrict__ x, float* __restrict__ mean,
                        float* __restrict__ rstd) {
  int s = blockIdx.x * 256 + threadIdx.x;        // 32768 sites
  int b = s >> 12, r = s & 4095;
  const float* px = x + (size_t)b * CCH * 4096 + r;
  float sum = 0.f, sq = 0.f;
  for (int c = 0; c < CCH; ++c) { float v = px[(size_t)c * 4096]; sum += v; sq += v * v; }
  float m = sum * (1.0f / 256.0f);
  float var = sq * (1.0f / 256.0f) - m * m;
  mean[s] = m; rstd[s] = rsqrtf(var + 1e-5f);
}

// fx = LN1(x) mapped into per-direction sequence layout [nl, c]; f16 + f32 copies.
__global__ void k_gather(const float* __restrict__ x, const float* __restrict__ mean,
                         const float* __restrict__ rstd,
                         const float* __restrict__ w1, const float* __restrict__ b1,
                         f16* __restrict__ fx16, float* __restrict__ fxf, int dir) {
  int t = blockIdx.x * 256 + threadIdx.x;        // over B*C*H*W
  int w = t & 63, h = (t >> 6) & 63, c = (t >> 12) & 255, b = t >> 20;
  int site = b * 4096 + h * 64 + w;
  float f = (x[t] - mean[site]) * rstd[site] * w1[c] + b1[c];
  int nl;
  if (dir == 0)      nl = (b * 64 + h) * 64 + w;
  else if (dir == 1) nl = (b * 64 + h) * 64 + (63 - w);
  else if (dir == 2) nl = (b * 64 + w) * 64 + h;
  else               nl = (b * 64 + w) * 64 + (63 - h);
  size_t o = (size_t)nl * 256 + c;
  fx16[o] = (f16)f; fxf[o] = f;
}

// ---------------- WMMA GEMM: C[M,N] = A[M,K]f16 * Bt[N,K]f16 ----------------
// Wave: 16(M) x 64(N) strip (4 accumulators). Block: 8 waves -> 128x64 tile.
// Inner loop is software-pipelined (double-buffered fragments) so each group of
// 4 WMMAs overlaps with the 10 global_load_b128 of the next K-slice.
// Epilogue modes:
//  0: plain       -> Cf[row,col] (+addsrc), optional f16 copy Ch
//  1: ssm post    -> Cf = gelu(acc + fx*aux[col]) + fx        (fx = addsrc)
//  2: dir scatter -> Ch(yall)[site*1024 + dir*256 + col] = f16(acc + addsrc)
//  3: NCHW store  -> Cf(out)[b, col, h, w] = acc  (row = site)
union Frag { v16h v; uint4 u[2]; };

__device__ __forceinline__ void load_a(Frag& f, const f16* arow, int k, int half) {
  f.u[0] = *(const uint4*)(arow + k + half * 8);
  f.u[1] = *(const uint4*)(arow + k + 16 + half * 8);
}
__device__ __forceinline__ void load_b(Frag& f, const f16* brow, int k, int half) {
  const f16* p = brow + k + half * 8;
  f.u[0] = *(const uint4*)(p);
  f.u[1] = *(const uint4*)(p + 16);
}

__global__ __launch_bounds__(256)
void k_gemm(const f16* __restrict__ A, const f16* __restrict__ Bt,
            float* __restrict__ Cf, f16* __restrict__ Ch,
            const float* __restrict__ addsrc, const float* __restrict__ aux,
            int M, int N, int K, int mode, int dir) {
  const int lane = threadIdx.x & 31;
  const int wv   = threadIdx.x >> 5;
  const int mr   = lane & 15;
  const int half = lane >> 4;
  const int m0   = blockIdx.x * 128 + wv * 16;
  const int n0   = blockIdx.y * 64;

  v8f acc[4] = {};
  const f16* arow = A + (size_t)(m0 + mr) * K;
  const f16* br0  = Bt + (size_t)(n0 + mr) * K;
  const f16* br1  = br0 + (size_t)16 * K;
  const f16* br2  = br0 + (size_t)32 * K;
  const f16* br3  = br0 + (size_t)48 * K;

  Frag a0, a1, b00, b01, b02, b03, b10, b11, b12, b13;

  // prologue: fetch K-slice 0
  load_a(a0, arow, 0, half);
  load_b(b00, br0, 0, half); load_b(b01, br1, 0, half);
  load_b(b02, br2, 0, half); load_b(b03, br3, 0, half);

  for (int k = 0; k < K; k += 64) {          // K is a multiple of 64
    // prefetch K-slice k+32 while slice k computes
    load_a(a1, arow, k + 32, half);
    load_b(b10, br0, k + 32, half); load_b(b11, br1, k + 32, half);
    load_b(b12, br2, k + 32, half); load_b(b13, br3, k + 32, half);

    acc[0] = __builtin_amdgcn_wmma_f32_16x16x32_f16(false, a0.v, false, b00.v, (short)0, acc[0], false, false);
    acc[1] = __builtin_amdgcn_wmma_f32_16x16x32_f16(false, a0.v, false, b01.v, (short)0, acc[1], false, false);
    acc[2] = __builtin_amdgcn_wmma_f32_16x16x32_f16(false, a0.v, false, b02.v, (short)0, acc[2], false, false);
    acc[3] = __builtin_amdgcn_wmma_f32_16x16x32_f16(false, a0.v, false, b03.v, (short)0, acc[3], false, false);

    if (k + 64 < K) {                        // prefetch K-slice k+64
      load_a(a0, arow, k + 64, half);
      load_b(b00, br0, k + 64, half); load_b(b01, br1, k + 64, half);
      load_b(b02, br2, k + 64, half); load_b(b03, br3, k + 64, half);
    }

    acc[0] = __builtin_amdgcn_wmma_f32_16x16x32_f16(false, a1.v, false, b10.v, (short)0, acc[0], false, false);
    acc[1] = __builtin_amdgcn_wmma_f32_16x16x32_f16(false, a1.v, false, b11.v, (short)0, acc[1], false, false);
    acc[2] = __builtin_amdgcn_wmma_f32_16x16x32_f16(false, a1.v, false, b12.v, (short)0, acc[2], false, false);
    acc[3] = __builtin_amdgcn_wmma_f32_16x16x32_f16(false, a1.v, false, b13.v, (short)0, acc[3], false, false);
  }

#pragma unroll
  for (int t = 0; t < 4; ++t) {
#pragma unroll
    for (int r = 0; r < 8; ++r) {
      int row = m0 + half * 8 + r;
      int col = n0 + t * 16 + mr;
      size_t idx = (size_t)row * N + col;
      float v = acc[t][r];
      if (mode == 0) {
        if (addsrc) v += addsrc[idx];
        if (Cf) Cf[idx] = v;
        if (Ch) Ch[idx] = (f16)v;
      } else if (mode == 1) {
        float fx = addsrc[idx];
        float v0 = v + fx * aux[col];
        float g = 0.5f * v0 * (1.0f + erff(v0 * 0.70710678118654752f));
        Cf[idx] = g + fx;
      } else if (mode == 2) {
        float v2 = v + addsrc[idx];
        int l = row & 63, n = row >> 6;
        int b = n >> 6, q = n & 63;
        int h, w;
        if (dir == 0)      { h = q; w = l; }
        else if (dir == 1) { h = q; w = 63 - l; }
        else if (dir == 2) { w = q; h = l; }
        else               { w = q; h = 63 - l; }
        int site = b * 4096 + h * 64 + w;
        Ch[(size_t)site * 1024 + dir * 256 + col] = (f16)v2;
      } else { // mode 3: row = site -> out[b, col, h, w]
        int b = row >> 12, hw = row & 4095;
        Cf[(size_t)b * (CCH * 4096) + (size_t)col * 4096 + hw] = v;
      }
    }
  }
}

// Diagonal complex scan: x_l = lam_bar * x_{l-1} + Bu_l ; one lane per (seq, state).
__global__ void k_scan(const f16* __restrict__ bu, f16* __restrict__ xs,
                       const float* __restrict__ lr_, const float* __restrict__ li_) {
  int t = blockIdx.x * 256 + threadIdx.x;        // 512*256
  int n = t >> 8, p = t & 255;
  float lr = lr_[p], li = li_[p];
  float xr = 0.f, xi = 0.f;
  size_t base = (size_t)n * 64 * 512 + p;
  for (int l = 0; l < LL; ++l) {
    size_t idx = base + (size_t)l * 512;
    float br = (float)bu[idx], bi = (float)bu[idx + 256];
    float nr = lr * xr - li * xi + br;
    float ni = lr * xi + li * xr + bi;
    xr = nr; xi = ni;
    xs[idx] = (f16)xr; xs[idx + 256] = (f16)xi;
  }
}

// LN2: one block per row, LDS tree reduction, write f16+f32 normalized rows.
__global__ __launch_bounds__(256)
void k_ln2(const float* __restrict__ y, const float* __restrict__ w2,
           const float* __restrict__ b2, f16* __restrict__ fx2h,
           float* __restrict__ fx2f) {
  __shared__ float sbuf[256];
  int row = blockIdx.x, c = threadIdx.x;
  size_t o = (size_t)row * 256 + c;
  float v = y[o];
  sbuf[c] = v; __syncthreads();
  for (int s = 128; s > 0; s >>= 1) { if (c < s) sbuf[c] += sbuf[c + s]; __syncthreads(); }
  float m = sbuf[0] * (1.0f / 256.0f); __syncthreads();
  float dv = v - m;
  sbuf[c] = dv * dv; __syncthreads();
  for (int s = 128; s > 0; s >>= 1) { if (c < s) sbuf[c] += sbuf[c + s]; __syncthreads(); }
  float var = sbuf[0] * (1.0f / 256.0f);
  float f = dv * rsqrtf(var + 1e-5f) * w2[c] + b2[c];
  fx2h[o] = (f16)f; fx2f[o] = f;
}

extern "C" void kernel_launch(void* const* d_in, const int* in_sizes, int n_in,
                              void* d_out, int out_size, void* d_ws, size_t ws_size,
                              hipStream_t stream) {
  (void)in_sizes; (void)n_in; (void)out_size; (void)ws_size;
  const float* x        = (const float*)d_in[0];
  const float* ln1_w    = (const float*)d_in[2];
  const float* ln1_b    = (const float*)d_in[3];
  const float* lam_re   = (const float*)d_in[4];
  const float* lam_im   = (const float*)d_in[5];
  const float* B_re     = (const float*)d_in[6];
  const float* B_im     = (const float*)d_in[7];
  const float* C_re     = (const float*)d_in[8];
  const float* C_im     = (const float*)d_in[9];
  const float* Dvec     = (const float*)d_in[10];
  const float* log_step = (const float*)d_in[11];
  const float* ln2_w    = (const float*)d_in[12];
  const float* ln2_b    = (const float*)d_in[13];
  const float* w_enc    = (const float*)d_in[14];
  const float* w_dec    = (const float*)d_in[15];
  const float* w_fuse   = (const float*)d_in[16];
  float* out = (float*)d_out;

  char* ws = (char*)d_ws;
  float* lbr  = (float*)(ws + OFF_LAMBR);
  float* lbi  = (float*)(ws + OFF_LAMBI);
  float* cfr  = (float*)(ws + OFF_COEFR);
  float* cfi  = (float*)(ws + OFF_COEFI);
  float* cmsk = (float*)(ws + OFF_CMASK);
  float* mean = (float*)(ws + OFF_MEAN);
  float* rstd = (float*)(ws + OFF_RSTD);
  f16*   wbu  = (f16*)  (ws + OFF_WBU);
  f16*   wc   = (f16*)  (ws + OFF_WC);
  f16*   wenc = (f16*)  (ws + OFF_WENC);
  f16*   wdec = (f16*)  (ws + OFF_WDEC);
  f16*   wfus = (f16*)  (ws + OFF_WFUSE);
  f16*   fx16 = (f16*)  (ws + OFF_FX16);   // also fx2 f16
  float* fxf  = (float*)(ws + OFF_FXF);    // also fx2 f32
  f16*   bu16 = (f16*)  (ws + OFF_BU16);
  f16*   xs16 = (f16*)  (ws + OFF_XS16);   // also hff16
  float* ysf  = (float*)(ws + OFF_YS);     // y (post-gelu)
  f16*   yall = (f16*)  (ws + OFF_YALL);

  // ---- parameter prep ----
  k_disc<<<4, 256, 0, stream>>>(lam_re, lam_im, log_step, lbr, lbi, cfr, cfi, cmsk);
  k_wbu <<<1024, 256, 0, stream>>>(B_re, B_im, cfr, cfi, wbu);
  k_wc  <<<1024, 256, 0, stream>>>(C_re, C_im, cmsk, wc);
  k_cvt16<<<1024, 256, 0, stream>>>(w_enc,  wenc);
  k_cvt16<<<1024, 256, 0, stream>>>(w_dec,  wdec);
  k_cvt16<<<1024, 256, 0, stream>>>(w_fuse, wfus);
  k_stats<<<SITES / 256, 256, 0, stream>>>(x, mean, rstd);

  const int GELT = (NL * CCH) / 256;
  for (int d = 0; d < 4; ++d) {
    k_gather<<<GELT, 256, 0, stream>>>(x, mean, rstd, ln1_w + d * 256, ln1_b + d * 256,
                                       fx16, fxf, d);
    // Bu (re||im) = fx @ B_bar^T : [32768,256] x [256,512], f16 out
    k_gemm<<<dim3(NL / 128, 512 / 64), 256, 0, stream>>>(
        fx16, wbu + (size_t)d * 131072, nullptr, bu16, nullptr, nullptr,
        NL, 512, 256, 0, d);
    k_scan<<<512, 256, 0, stream>>>(bu16, xs16, lbr + d * 256, lbi + d * 256);
    // y = gelu([xr||xi] @ Wc^T + fx*D) + fx  (mode 1 epilogue)
    k_gemm<<<dim3(NL / 128, 256 / 64), 256, 0, stream>>>(
        xs16, wc + (size_t)d * 131072, ysf, nullptr, fxf, Dvec + d * 256,
        NL, 256, 512, 1, d);
    k_ln2<<<NL, 256, 0, stream>>>(ysf, ln2_w + d * 256, ln2_b + d * 256, fx16, fxf);
    // hff = fx2 @ w_enc^T, f16 out
    k_gemm<<<dim3(NL / 128, 256 / 64), 256, 0, stream>>>(
        fx16, wenc + (size_t)d * 65536, nullptr, xs16, nullptr, nullptr,
        NL, 256, 256, 0, d);
    // Yall[site, d*256+c] = f16(hff @ w_dec^T + fx2)  (mode 2 epilogue)
    k_gemm<<<dim3(NL / 128, 256 / 64), 256, 0, stream>>>(
        xs16, wdec + (size_t)d * 65536, nullptr, yall, fxf, nullptr,
        NL, 256, 256, 2, d);
  }

  // out[b,c,h,w] = Yall[site,1024] @ w_fuse^T  (mode 3 epilogue, K=1024)
  k_gemm<<<dim3(NL / 128, 256 / 64), 256, 0, stream>>>(
      yall, wfus, out, nullptr, nullptr, nullptr,
      NL, 256, 1024, 3, 0);
}
